// FuseAttention_15083925143641
// MI455X (gfx1250) — matrix operand order repaired and checked by reading, not compiled, not run on hardware
//
#include <hip/hip_runtime.h>
#include <hip/hip_bf16.h>
#include <stdint.h>
#include <stddef.h>

typedef __bf16 bf16_t;
typedef __attribute__((ext_vector_type(16))) __bf16 v16bf;
typedef __attribute__((ext_vector_type(8)))  float   v8f;
typedef __attribute__((ext_vector_type(4)))  float   v4f;

#define KDIM 128
#define TX 4
#define DLEN 512
#define BTOT 32768
#define INV_SQRT2f 0.70710678118654752440f

// workspace layout (bf16 elements): Wq | Wk | Wv | conv_w
#define WS_WQ 0
#define WS_WK (128*128)
#define WS_WV (2*128*128)
#define WS_CW (3*128*128)
#define WS_TOT (3*128*128 + 128*256)

// ---------------------------------------------------------------------------
// Weight fp32 -> bf16 conversion (one-shot, tiny)
// ---------------------------------------------------------------------------
__global__ __launch_bounds__(256) void cvt_weights_kernel(
    const float* __restrict__ wq, const float* __restrict__ wk,
    const float* __restrict__ wv, const float* __restrict__ cw,
    bf16_t* __restrict__ ws) {
  int i = blockIdx.x * blockDim.x + threadIdx.x;
  if (i < WS_WK)            ws[i] = (bf16_t)wq[i];
  else if (i < WS_WV)       ws[i] = (bf16_t)wk[i - WS_WK];
  else if (i < WS_CW)       ws[i] = (bf16_t)wv[i - WS_WV];
  else if (i < WS_TOT)      ws[i] = (bf16_t)cw[i - WS_CW];
}

// ---------------------------------------------------------------------------
// Helpers
// ---------------------------------------------------------------------------
// Load a 16-element bf16 fragment as two 16B chunks (A: K=kb..kb+15 contiguous,
// B: K=kb2..kb2+15 contiguous -> p1 = p0+8).
__device__ __forceinline__ v16bf ld_frag(const bf16_t* p0, const bf16_t* p1) {
  union { v16bf v; v4f f[2]; } u;
  u.f[0] = *(const v4f*)p0;
  u.f[1] = *(const v4f*)p1;
  return u.v;
}

// Async bulk copy (bf16, nelem multiple of 8*blockDim) global -> LDS using the
// CDNA5 async-to-LDS path, then drain this wave's ASYNCcnt.
__device__ __forceinline__ void lds_fill_async(bf16_t* dst, const bf16_t* __restrict__ src,
                                               int nelem) {
  uint32_t ldsOff = (uint32_t)(uintptr_t)dst;   // low 32 bits of flat LDS addr = LDS offset
  for (int i = (int)threadIdx.x * 8; i < nelem; i += (int)blockDim.x * 8) {
    uint32_t d = ldsOff + (uint32_t)(i * 2);
    uint64_t a = (uint64_t)(uintptr_t)(src + i);
    asm volatile("global_load_async_to_lds_b128 %0, %1, off" :: "v"(d), "v"(a) : "memory");
  }
  asm volatile("s_wait_asynccnt 0" ::: "memory");
}

__device__ __forceinline__ float bn_lrelu(float acc, float a, float c) {
  float t = fmaf(acc, a, c);
  return t > 0.f ? t : 0.01f * t;
}

#define WMMA_BF16(A, B, C) \
  __builtin_amdgcn_wmma_f32_16x16x32_bf16(false, (A), false, (B), (short)0, (C), false, false)

// ---------------------------------------------------------------------------
// Fused attention + IDWT + conv1x1 + BN + LeakyReLU
// block = 64 threads (2 waves), each wave owns a 16-row batch tile
// ---------------------------------------------------------------------------
struct alignas(16) PW {
  bf16_t vu[TX][16][KDIM];   // V tiles, Haar-combined in place (16 KB)
  bf16_t xb[TX][16][KDIM];   // x staged bf16 (16 KB)
  bf16_t yb[16][KDIM];       // y staged bf16 (4 KB)
  float  q[16][KDIM];        // Q tile fp32 (8 KB)
  float  dw[TX][16];         // dots -> softmax weights
};

__global__ __launch_bounds__(64) void fused_attn_idwt_conv_kernel(
    const float* __restrict__ x, const float* __restrict__ y,
    const bf16_t* __restrict__ ws,
    const float* __restrict__ conv_b, const float* __restrict__ gamma,
    const float* __restrict__ beta,  const float* __restrict__ mean,
    const float* __restrict__ var,   float* __restrict__ out) {
  __shared__ bf16_t sW[128 * 256];        // one weight matrix at a time (64 KB)
  __shared__ float fA[KDIM], fC[KDIM];    // folded BN/conv_b constants
  __shared__ PW pw[2];

  const int wave = threadIdx.x >> 5;
  const int lane = threadIdx.x & 31;
  const int h    = lane >> 4;             // wave half (0/1)
  const int ln   = lane & 15;
  const int tileBase = blockIdx.x * 32 + wave * 16;
  PW& P = pw[wave];

  // ---- phase 0: fold BN constants; stage y and x tiles as bf16 ----
  for (int o = threadIdx.x; o < KDIM; o += blockDim.x) {
    float sc = gamma[o] * rsqrtf(var[o] + 1e-5f);
    fA[o] = INV_SQRT2f * sc;
    fC[o] = (conv_b[o] - mean[o]) * sc + beta[o];
  }
  for (int i = lane; i < 16 * (KDIM / 4); i += 32) {
    int r = i >> 5, c4 = (i & 31) << 2;
    v4f f = *(const v4f*)(y + (size_t)(tileBase + r) * KDIM + c4);
    bf16_t* d = &P.yb[r][c4];
    d[0] = (bf16_t)f.x; d[1] = (bf16_t)f.y; d[2] = (bf16_t)f.z; d[3] = (bf16_t)f.w;
  }
  for (int i = lane; i < TX * 16 * (KDIM / 4); i += 32) {
    int t = i >> 9, r = (i >> 5) & 15, c4 = (i & 31) << 2;
    v4f f = *(const v4f*)(x + ((size_t)(tileBase + r) * TX + t) * KDIM + c4);
    bf16_t* d = &P.xb[t][r][c4];
    d[0] = (bf16_t)f.x; d[1] = (bf16_t)f.y; d[2] = (bf16_t)f.z; d[3] = (bf16_t)f.w;
  }
  __syncthreads();

  // ---- phase 1: Q = yb * Wq^T  (16x128) ----
  lds_fill_async(sW, ws + WS_WQ, KDIM * KDIM);
  __syncthreads();
  for (int nt = 0; nt < 8; ++nt) {
    v8f acc = {};
#pragma unroll
    for (int ks = 0; ks < 4; ++ks) {
      int kb = (h << 4) + (ks << 5);
      v16bf a = ld_frag(&P.yb[ln][kb], &P.yb[ln][kb + 8]);
      const bf16_t* wr = sW + (nt * 16 + ln) * KDIM + (h << 4) + (ks << 5);
      v16bf b = ld_frag(wr, wr + 8);
      acc = WMMA_BF16(a, b, acc);
    }
#pragma unroll
    for (int r = 0; r < 8; ++r) P.q[h * 8 + r][nt * 16 + ln] = acc[r];
  }
  __syncthreads();

  // ---- phase 2: keys + q.keys dots ----
  lds_fill_async(sW, ws + WS_WK, KDIM * KDIM);
  __syncthreads();
  const float dscale = 0.08838834764831844f;  // 1/sqrt(128)
  for (int t = 0; t < TX; ++t) {
    float pr[8] = {0, 0, 0, 0, 0, 0, 0, 0};
    for (int nt = 0; nt < 8; ++nt) {
      v8f acc = {};
#pragma unroll
      for (int ks = 0; ks < 4; ++ks) {
        int kb = (h << 4) + (ks << 5);
        v16bf a = ld_frag(&P.xb[t][ln][kb], &P.xb[t][ln][kb + 8]);
        const bf16_t* wr = sW + (nt * 16 + ln) * KDIM + (h << 4) + (ks << 5);
        v16bf b = ld_frag(wr, wr + 8);
        acc = WMMA_BF16(a, b, acc);
      }
#pragma unroll
      for (int r = 0; r < 8; ++r) pr[r] += acc[r] * P.q[h * 8 + r][nt * 16 + ln];
    }
#pragma unroll
    for (int r = 0; r < 8; ++r) {  // reduce across the 16 lanes of this half
      float v = pr[r];
      v += __shfl_xor(v, 1); v += __shfl_xor(v, 2);
      v += __shfl_xor(v, 4); v += __shfl_xor(v, 8);
      pr[r] = v;
    }
    if (ln == 0) {
#pragma unroll
      for (int r = 0; r < 8; ++r) P.dw[t][h * 8 + r] = pr[r] * dscale;
    }
  }
  __syncthreads();

  // ---- softmax over t (lanes 0..15 each own one batch row) ----
  if (h == 0) {
    float d0 = P.dw[0][ln], d1 = P.dw[1][ln], d2 = P.dw[2][ln], d3 = P.dw[3][ln];
    float mx = fmaxf(fmaxf(d0, d1), fmaxf(d2, d3));
    float e0 = expf(d0 - mx), e1 = expf(d1 - mx), e2 = expf(d2 - mx), e3 = expf(d3 - mx);
    float inv = 1.f / (e0 + e1 + e2 + e3);
    P.dw[0][ln] = e0 * inv; P.dw[1][ln] = e1 * inv;
    P.dw[2][ln] = e2 * inv; P.dw[3][ln] = e3 * inv;
  }
  __syncthreads();

  // ---- phase 3: V = xb * Wv^T, weighted by softmax, stored bf16 ----
  lds_fill_async(sW, ws + WS_WV, KDIM * KDIM);
  __syncthreads();
  for (int t = 0; t < TX; ++t) {
    for (int nt = 0; nt < 8; ++nt) {
      v8f acc = {};
#pragma unroll
      for (int ks = 0; ks < 4; ++ks) {
        int kb = (h << 4) + (ks << 5);
        v16bf a = ld_frag(&P.xb[t][ln][kb], &P.xb[t][ln][kb + 8]);
        const bf16_t* wr = sW + (nt * 16 + ln) * KDIM + (h << 4) + (ks << 5);
        v16bf b = ld_frag(wr, wr + 8);
        acc = WMMA_BF16(a, b, acc);
      }
#pragma unroll
      for (int r = 0; r < 8; ++r) {
        float wt = P.dw[t][h * 8 + r];
        P.vu[t][h * 8 + r][nt * 16 + ln] = (bf16_t)(acc[r] * wt);
      }
    }
  }
  __syncthreads();

  // ---- Haar pre-combine in place: vu0/2 = sums (even), vu1/3 = diffs (odd) ----
  for (int i = lane; i < 16 * KDIM; i += 32) {
    int m = i >> 7, c = i & 127;
    float a0 = (float)P.vu[0][m][c], a1 = (float)P.vu[1][m][c];
    float a2 = (float)P.vu[2][m][c], a3 = (float)P.vu[3][m][c];
    P.vu[0][m][c] = (bf16_t)(a0 + a1); P.vu[1][m][c] = (bf16_t)(a0 - a1);
    P.vu[2][m][c] = (bf16_t)(a2 + a3); P.vu[3][m][c] = (bf16_t)(a2 - a3);
  }
  __syncthreads();

  // ---- phase 4: conv (K=256) for even+odd, BN + LeakyReLU, coalesced store ----
  lds_fill_async(sW, ws + WS_CW, KDIM * 256);
  __syncthreads();
  const int g = tileBase / DLEN;
  const int dbase = (tileBase % DLEN) + (h << 3);
  float* outg = out + (size_t)g * (KDIM * 1024);
  for (int nt = 0; nt < 8; ++nt) {
    v8f accE = {}, accO = {};
#pragma unroll
    for (int ks = 0; ks < 8; ++ks) {
      int tsE = (ks < 4) ? 0 : 2;                 // low/high Haar channel block
      int kb = (h << 4) + ((ks & 3) << 5);        // K offset within 128-block
      v16bf aE = ld_frag(&P.vu[tsE][ln][kb],     &P.vu[tsE][ln][kb + 8]);
      v16bf aO = ld_frag(&P.vu[tsE + 1][ln][kb], &P.vu[tsE + 1][ln][kb + 8]);
      const bf16_t* wr = sW + (nt * 16 + ln) * 256 + (h << 4) + (ks << 5);
      v16bf b = ld_frag(wr, wr + 8);              // shared between even/odd
      accE = WMMA_BF16(aE, b, accE);
      accO = WMMA_BF16(aO, b, accO);
    }
    int o = nt * 16 + ln;
    float a = fA[o], c = fC[o];
    float* dst = outg + (size_t)o * 1024 + 2 * dbase;   // 16 contiguous floats / lane
#pragma unroll
    for (int qi = 0; qi < 4; ++qi) {
      v4f o4;
      o4.x = bn_lrelu(accE[2 * qi],     a, c);
      o4.y = bn_lrelu(accO[2 * qi],     a, c);
      o4.z = bn_lrelu(accE[2 * qi + 1], a, c);
      o4.w = bn_lrelu(accO[2 * qi + 1], a, c);
      *(v4f*)(dst + (qi << 2)) = o4;
    }
  }
}

// ---------------------------------------------------------------------------
extern "C" void kernel_launch(void* const* d_in, const int* in_sizes, int n_in,
                              void* d_out, int out_size, void* d_ws, size_t ws_size,
                              hipStream_t stream) {
  (void)in_sizes; (void)n_in; (void)out_size; (void)ws_size;
  const float* x      = (const float*)d_in[0];
  const float* y      = (const float*)d_in[1];
  const float* Wq     = (const float*)d_in[2];
  const float* Wk     = (const float*)d_in[3];
  const float* Wv     = (const float*)d_in[4];
  const float* Cw     = (const float*)d_in[5];
  const float* conv_b = (const float*)d_in[6];
  const float* gamma  = (const float*)d_in[7];
  const float* beta   = (const float*)d_in[8];
  const float* mean   = (const float*)d_in[9];
  const float* var    = (const float*)d_in[10];
  bf16_t* ws = (bf16_t*)d_ws;
  float* out = (float*)d_out;

  cvt_weights_kernel<<<(WS_TOT + 255) / 256, 256, 0, stream>>>(Wq, Wk, Wv, Cw, ws);
  fused_attn_idwt_conv_kernel<<<BTOT / 32, 64, 0, stream>>>(
      x, y, ws, conv_b, gamma, beta, mean, var, out);
}
